// GraphCNN_41686952575135
// MI455X (gfx1250) — matrix operand order repaired and checked by reading, not compiled, not run on hardware
//
#include <hip/hip_runtime.h>
#include <hip/hip_bf16.h>

// ---------------------------------------------------------------------------
// GraphCNN on gfx1250: f16 WMMA GEMMs (f32 accumulate), LDS-staged weights,
// f32 sparse path with hardware float atomics.
// ---------------------------------------------------------------------------

#define N_NODES  50000
#define IN_DIM   128
#define HID_DIM  256
#define OUT_DIM  128
#define MAX_DEPTH 8
#define N_EDGES  600000

typedef _Float16 h8  __attribute__((ext_vector_type(8)));
typedef _Float16 h16 __attribute__((ext_vector_type(16)));
typedef float    v8f __attribute__((ext_vector_type(8)));

// ---------------- A-fragment loader (16x32 f16 A, ISA VGPR layout) ----------
// lane = 16*h + m holds row m: a[0..7]  = A[row][k0+8h    .. +7],
//                              a[8..15] = A[row][k0+16+8h .. +7]
template <bool F32A>
__device__ inline h16 load_a_frag(const void* __restrict__ A, int row, int K,
                                  int k0, int h) {
  h16 a;
  if constexpr (F32A) {
    const float* ap = (const float*)A + (size_t)row * K + k0 + 8 * h;
#pragma unroll
    for (int i = 0; i < 8; ++i) a[i] = (_Float16)ap[i];
#pragma unroll
    for (int i = 0; i < 8; ++i) a[8 + i] = (_Float16)ap[16 + i];
  } else {
    const _Float16* ap = (const _Float16*)A + (size_t)row * K + k0 + 8 * h;
    h8 lo = *(const h8*)ap;
    h8 hi = *(const h8*)(ap + 16);
#pragma unroll
    for (int i = 0; i < 8; ++i) { a[i] = lo[i]; a[8 + i] = hi[i]; }
  }
  return a;
}

// ---------------- fused GEMM: out = leaky_relu(A @ W + b) -------------------
// OUT_MODE: 0 = f16 out, 1 = f32 out, 2 = both, 3 = residual: both, where
//           v = xold + mask[row]*leaky(acc+b)
// Block: 256 threads = 8 waves. Block tile = 128 rows x 64 cols.
// Wave w -> rows [bx*128 + 16w, +16), all 64 cols as 4 WMMA tiles sharing
// one A fragment per K-step. B slice (64 cols x K) staged in LDS once.
template <bool F32A, int OUT_MODE>
__global__ __launch_bounds__(256) void gemm_wmma(
    const void* __restrict__ A, const _Float16* __restrict__ Wt,
    const float* __restrict__ bias, int N, int K, int M,
    _Float16* __restrict__ out16, float* __restrict__ out32,
    const float* __restrict__ xold, const float* __restrict__ mask) {
  constexpr int KMAX = 256;
  constexpr int KPAD = 8;  // +8 halfs -> dword stride % 64 banks == 4
  __shared__ __align__(16) _Float16 Bsh[64 * (KMAX + KPAD)];
  const int KP = K + KPAD;

  const int lane = threadIdx.x & 31;
  const int wave = threadIdx.x >> 5;
  const int mn   = lane & 15;   // row within A tile / col within B tile
  const int h    = lane >> 4;   // half-wave selector
  const int rowbase = blockIdx.x * 128 + wave * 16;
  const int colbase = blockIdx.y * 64;
  int rowA = rowbase + mn;
  if (rowA > N - 1) rowA = N - 1;  // clamp (keeps EXEC all-ones for WMMA)

  // ---- cooperative stage: Wt[colbase+c][0..K) -> Bsh[c*KP + 0..K) ----
  {
    const int kc = K / 8;            // 16B chunks per column
    const int chunks = 64 * kc;
    for (int c = threadIdx.x; c < chunks; c += 256) {
      const int col = c / kc;
      const int ko  = (c - col * kc) * 8;
      *(h8*)&Bsh[col * KP + ko] =
          *(const h8*)(Wt + (size_t)(colbase + col) * K + ko);
    }
  }
  __syncthreads();

  v8f acc[4] = {v8f{}, v8f{}, v8f{}, v8f{}};
  for (int k0 = 0; k0 < K; k0 += 32) {
    const h16 a = load_a_frag<F32A>(A, rowA, K, k0, h);
#pragma unroll
    for (int nt = 0; nt < 4; ++nt) {
      // B frag: lane = 16h + n holds col n of tile, K = k0+8h .. +7, k0+16+8h .. +7
      const _Float16* bp = &Bsh[(nt * 16 + mn) * KP + k0 + 8 * h];
      h8 lo = *(const h8*)bp;        // ds_load_b128
      h8 hi = *(const h8*)(bp + 16);
      h16 b;
#pragma unroll
      for (int i = 0; i < 8; ++i) { b[i] = lo[i]; b[8 + i] = hi[i]; }
      acc[nt] = __builtin_amdgcn_wmma_f32_16x16x32_f16(
          false, a, false, b, (short)0, acc[nt], false, false);
    }
  }

  // Epilogue. C/D layout: acc[nt][r] -> row = rowbase + r + 8h,
  //                                     col = colbase + nt*16 + mn
#pragma unroll
  for (int nt = 0; nt < 4; ++nt) {
    const int col = colbase + nt * 16 + mn;
    const float bv = bias[col];
#pragma unroll
    for (int r = 0; r < 8; ++r) {
      const int orow = rowbase + r + 8 * h;
      if (orow >= N) continue;
      float v = acc[nt][r] + bv;
      v = v > 0.f ? v : 0.01f * v;   // leaky_relu(0.01)
      if constexpr (OUT_MODE == 3) {
        v = xold[(size_t)orow * M + col] + mask[orow] * v;
      }
      if constexpr (OUT_MODE == 0 || OUT_MODE == 2 || OUT_MODE == 3) {
        out16[(size_t)orow * M + col] = (_Float16)v;
      }
      if constexpr (OUT_MODE == 1 || OUT_MODE == 2 || OUT_MODE == 3) {
        out32[(size_t)orow * M + col] = v;
      }
    }
  }
}

// ---------------- weight cast + transpose: W[K][M] f32 -> Wt[M][K] f16 ------
__global__ __launch_bounds__(256) void cast_transpose(
    const float* __restrict__ W, _Float16* __restrict__ Wt, int K, int M) {
  int idx = blockIdx.x * blockDim.x + threadIdx.x;
  if (idx >= K * M) return;
  int k = idx / M, m = idx % M;
  Wt[(size_t)m * K + k] = (_Float16)W[idx];
}

// ---------------- sparse step: agg[src[e]] += val[e] * y[dst[e]] ------------
// One wave per edge; each lane owns 4 of the 128 features (float4 load).
__global__ __launch_bounds__(256) void edge_scatter(
    const float* __restrict__ y, const int* __restrict__ src,
    const int* __restrict__ dst, const float* __restrict__ val,
    float* __restrict__ agg, int nedges) {
  const int e = blockIdx.x * 8 + (threadIdx.x >> 5);
  if (e >= nedges) return;
  const int lane = threadIdx.x & 31;
  const int s = src[e];
  const int d = dst[e];
  const float v = val[e];
  const float4 q = *(const float4*)(y + (size_t)d * OUT_DIM + lane * 4);
  float* o = agg + (size_t)s * OUT_DIM + lane * 4;
  unsafeAtomicAdd(o + 0, v * q.x);   // -> global_atomic_add_f32
  unsafeAtomicAdd(o + 1, v * q.y);
  unsafeAtomicAdd(o + 2, v * q.z);
  unsafeAtomicAdd(o + 3, v * q.w);
}

// ---------------------------------------------------------------------------
extern "C" void kernel_launch(void* const* d_in, const int* in_sizes, int n_in,
                              void* d_out, int out_size, void* d_ws,
                              size_t ws_size, hipStream_t stream) {
  (void)in_sizes; (void)n_in; (void)out_size; (void)ws_size;
  const float* node_inputs = (const float*)d_in[0];
  const int*   edge_src    = (const int*)d_in[1];
  const int*   edge_dst    = (const int*)d_in[2];
  const float* adj_vals    = (const float*)d_in[3];
  const float* masks       = (const float*)d_in[4];
  const float* W_h1 = (const float*)d_in[5],  *b_h1 = (const float*)d_in[6];
  const float* W_h2 = (const float*)d_in[7],  *b_h2 = (const float*)d_in[8];
  const float* W_h3 = (const float*)d_in[9],  *b_h3 = (const float*)d_in[10];
  const float* W_f1 = (const float*)d_in[11], *b_f1 = (const float*)d_in[12];
  const float* W_f2 = (const float*)d_in[13], *b_f2 = (const float*)d_in[14];
  const float* W_f3 = (const float*)d_in[15], *b_f3 = (const float*)d_in[16];

  const int N = N_NODES;
  char* ws = (char*)d_ws;
  size_t off = 0;
  auto carve = [&](size_t bytes) -> void* {
    void* p = ws + off;
    off += (bytes + 255) & ~(size_t)255;
    return p;
  };
  _Float16* Wh1t = (_Float16*)carve(IN_DIM  * HID_DIM * 2);
  _Float16* Wh2t = (_Float16*)carve(HID_DIM * HID_DIM * 2);
  _Float16* Wh3t = (_Float16*)carve(HID_DIM * OUT_DIM * 2);
  _Float16* Wf1t = (_Float16*)carve(OUT_DIM * HID_DIM * 2);
  _Float16* Wf2t = (_Float16*)carve(HID_DIM * HID_DIM * 2);
  _Float16* Wf3t = (_Float16*)carve(HID_DIM * OUT_DIM * 2);
  float*    x32  = (float*)   carve((size_t)N * OUT_DIM * 4);
  _Float16* x16  = (_Float16*)carve((size_t)N * OUT_DIM * 2);
  void*     buf1 = carve((size_t)N * HID_DIM * 2);  // h1 f16 OR y f32 (same bytes)
  _Float16* h2   = (_Float16*)carve((size_t)N * HID_DIM * 2);
  float*    agg  = (float*)   carve((size_t)N * OUT_DIM * 4);

  _Float16* h1f16 = (_Float16*)buf1;
  float*    yf32  = (float*)buf1;

  // ---- weights -> f16 transposed (tiny; once per call) ----
  auto castgrid = [](int n) { return dim3((n + 255) / 256); };
  cast_transpose<<<castgrid(IN_DIM * HID_DIM),  256, 0, stream>>>(W_h1, Wh1t, IN_DIM,  HID_DIM);
  cast_transpose<<<castgrid(HID_DIM * HID_DIM), 256, 0, stream>>>(W_h2, Wh2t, HID_DIM, HID_DIM);
  cast_transpose<<<castgrid(HID_DIM * OUT_DIM), 256, 0, stream>>>(W_h3, Wh3t, HID_DIM, OUT_DIM);
  cast_transpose<<<castgrid(OUT_DIM * HID_DIM), 256, 0, stream>>>(W_f1, Wf1t, OUT_DIM, HID_DIM);
  cast_transpose<<<castgrid(HID_DIM * HID_DIM), 256, 0, stream>>>(W_f2, Wf2t, HID_DIM, HID_DIM);
  cast_transpose<<<castgrid(HID_DIM * OUT_DIM), 256, 0, stream>>>(W_f3, Wf3t, HID_DIM, OUT_DIM);

  const dim3 blk(256);
  const int gx = (N + 127) / 128;
  const dim3 gridH(gx, HID_DIM / 64);   // M = 256 -> 4 col blocks
  const dim3 gridO(gx, OUT_DIM / 64);   // M = 128 -> 2 col blocks

  // ---- embedding MLP: x = mlp3(node_inputs, W_h*) ----
  gemm_wmma<true, 0><<<gridH, blk, 0, stream>>>(
      node_inputs, Wh1t, b_h1, N, IN_DIM, HID_DIM, h1f16, nullptr, nullptr, nullptr);
  gemm_wmma<false, 0><<<gridH, blk, 0, stream>>>(
      h1f16, Wh2t, b_h2, N, HID_DIM, HID_DIM, h2, nullptr, nullptr, nullptr);
  gemm_wmma<false, 2><<<gridO, blk, 0, stream>>>(
      h2, Wh3t, b_h3, N, HID_DIM, OUT_DIM, x16, x32, nullptr, nullptr);

  // ---- propagation depths ----
  for (int d = 0; d < MAX_DEPTH; ++d) {
    // y = mlp3(x, W_f*)
    gemm_wmma<false, 0><<<gridH, blk, 0, stream>>>(
        x16, Wf1t, b_f1, N, OUT_DIM, HID_DIM, h1f16, nullptr, nullptr, nullptr);
    gemm_wmma<false, 0><<<gridH, blk, 0, stream>>>(
        h1f16, Wf2t, b_f2, N, HID_DIM, HID_DIM, h2, nullptr, nullptr, nullptr);
    gemm_wmma<false, 1><<<gridO, blk, 0, stream>>>(
        h2, Wf3t, b_f3, N, HID_DIM, OUT_DIM, nullptr, yf32, nullptr, nullptr);

    // agg[i] = sum_{e: src=i} val_e * y[dst_e]
    hipMemsetAsync(agg, 0, (size_t)N * OUT_DIM * 4, stream);
    edge_scatter<<<dim3((N_EDGES + 7) / 8), blk, 0, stream>>>(
        yf32, edge_src + (size_t)d * N_EDGES, edge_dst + (size_t)d * N_EDGES,
        adj_vals + (size_t)d * N_EDGES, agg, N_EDGES);

    // y2 = mlp3(agg, W_f*);  x = x + mask * y2  (fused in last GEMM)
    gemm_wmma<true, 0><<<gridH, blk, 0, stream>>>(
        agg, Wf1t, b_f1, N, OUT_DIM, HID_DIM, h1f16, nullptr, nullptr, nullptr);
    gemm_wmma<false, 0><<<gridH, blk, 0, stream>>>(
        h1f16, Wf2t, b_f2, N, HID_DIM, HID_DIM, h2, nullptr, nullptr, nullptr);
    float* xout = (d == MAX_DEPTH - 1) ? (float*)d_out : x32;
    gemm_wmma<false, 3><<<gridO, blk, 0, stream>>>(
        h2, Wf3t, b_f3, N, HID_DIM, OUT_DIM, x16, xout, x32,
        masks + (size_t)d * N);
  }
}
